// FusionRetrModel_21423296873159
// MI455X (gfx1250) — matrix operand order repaired and checked by reading, not compiled, not run on hardware
//
#include <hip/hip_runtime.h>
#include <hip/hip_bf16.h>

typedef _Float16 half_t;
typedef _Float16 v8h  __attribute__((ext_vector_type(8)));
typedef _Float16 v16h __attribute__((ext_vector_type(16)));
typedef float    v8f  __attribute__((ext_vector_type(8)));
typedef float    f32x4 __attribute__((ext_vector_type(4)));
typedef int      v4i  __attribute__((ext_vector_type(4)));

#define B_    4
#define NL_   2
#define P_    100
#define TOK_  60
#define T_    6000          // P_*TOK_
#define D_    768
#define NT_   10
#define MTOT  (B_*T_)       // 24000
#define K1_   (2*D_)        // 1536  (X' = [Q, A*Q])
#define LDA   40            // LDS row stride in halves (pad 32 -> 40)

// ---------------- CDNA5 async global->LDS (ASYNCcnt) ----------------

#if defined(__has_builtin)
#if __has_builtin(__builtin_amdgcn_global_load_async_to_lds_b128) && \
    __has_builtin(__builtin_amdgcn_s_wait_asynccnt)
#define HAVE_ASYNC_LDS 1
#endif
#endif

typedef __attribute__((address_space(1))) v4i* gptr_v4i;
typedef __attribute__((address_space(3))) v4i* lptr_v4i;

static __device__ __forceinline__ void async_cp16(const half_t* g, half_t* l) {
#ifdef HAVE_ASYNC_LDS
  __builtin_amdgcn_global_load_async_to_lds_b128((gptr_v4i)g, (lptr_v4i)l, 0, 0);
#else
  *(v8h*)l = *(const v8h*)g;
#endif
}
static __device__ __forceinline__ void async_wait() {
#ifdef HAVE_ASYNC_LDS
  __builtin_amdgcn_s_wait_asynccnt(0);
#endif
}

// ---------------- WMMA helpers (CDNA5 wave32 layouts) ----------------

static __device__ __forceinline__ v8f wmma_f16(v16h a, v16h b, v8f c) {
  // v_wmma_f32_16x16x32_f16
  return __builtin_amdgcn_wmma_f32_16x16x32_f16(false, a, false, b, (short)0, c,
                                                false, false);
}

// A-matrix 16x32 f16 fragment from a row-major LDS tile (stride LDA halves).
// lane<16:  row=lane,    K = {0..7, 16..23}
// lane>=16: row=lane-16, K = {8..15, 24..31}
static __device__ __forceinline__ v16h load_a_frag(const half_t* As, int lane, int mrow) {
  int row = mrow + (lane & 15);
  int h0  = (lane >> 4) << 3;                  // 0 or 8
  v8h lo = *(const v8h*)(As + row * LDA + h0);
  v8h hi = *(const v8h*)(As + row * LDA + h0 + 16);
  return __builtin_shufflevector(lo, hi, 0,1,2,3,4,5,6,7,8,9,10,11,12,13,14,15);
}

// B-matrix 32x16 f16 fragment from an n-major LDS tile Bs[n][k] (stride LDA).
// lane<16: col=lane, K={0..7,16..23} ; lane>=16: col=lane-16, K={8..15,24..31}
static __device__ __forceinline__ v16h load_b_frag(const half_t* Bs, int lane, int ncol) {
  int col = ncol + (lane & 15);
  int k0  = (lane >> 4) << 3;                  // 0 or 8
  v8h lo = *(const v8h*)(Bs + col * LDA + k0);
  v8h hi = *(const v8h*)(Bs + col * LDA + k0 + 16);
  return __builtin_shufflevector(lo, hi, 0,1,2,3,4,5,6,7,8,9,10,11,12,13,14,15);
}

// convert 8 consecutive f32 (two float4 loads) to v8h
static __device__ __forceinline__ v8h cvt8(const float* p) {
  f32x4 a = *(const f32x4*)p;
  f32x4 b = *(const f32x4*)(p + 4);
  v8h r;
  r[0] = (half_t)a[0]; r[1] = (half_t)a[1]; r[2] = (half_t)a[2]; r[3] = (half_t)a[3];
  r[4] = (half_t)b[0]; r[5] = (half_t)b[1]; r[6] = (half_t)b[2]; r[7] = (half_t)b[3];
  return r;
}

// ---------------- K0: init TMAX to -inf, score to 0 ----------------

__global__ void k_init(float* __restrict__ TMAX, float* __restrict__ score) {
  int idx = blockIdx.x * blockDim.x + threadIdx.x;
  const int NTM = B_ * NT_ * TOK_ * D_;        // 1,843,200
  if (idx < NTM) TMAX[idx] = -__builtin_inff();
  int s = idx - NTM;
  if (s >= 0 && s < MTOT) score[s] = 0.0f;
}

// ---------------- KW: repack weights to f16, n-major (B operand) ----------------

__global__ void k_prep_w(const float* __restrict__ Wp, const float* __restrict__ Wt,
                         const float* __restrict__ Wf1,
                         half_t* __restrict__ WT16, half_t* __restrict__ WF16,
                         half_t* __restrict__ WA16) {
  int idx = blockIdx.x * blockDim.x + threadIdx.x;
  const int N1 = K1_ * K1_;                    // 2,359,296
  const int N2 = D_ * D_;
  if (idx < N1) {
    int n = idx / K1_, k = idx - n * K1_;
    float v = (n < D_) ? Wp[(size_t)n * (3 * D_) + D_ + k]
                       : Wt[(size_t)(n - D_) * (3 * D_) + D_ + k];
    WT16[idx] = (half_t)v;
  } else if (idx < N1 + N2) {
    int e = idx - N1; int n = e / D_, k = e - n * D_;
    WF16[e] = (half_t)Wf1[(size_t)n * (2 * D_) + k];
  } else if (idx < N1 + 2 * N2) {
    int e = idx - N1 - N2; int n = e / D_, k = e - n * D_;
    WA16[e] = (half_t)Wf1[(size_t)n * (2 * D_) + D_ + k];
  }
}

// ---------------- KC: per-batch constant terms ----------------
// cp[b][n] = bp[n] + sum_j Wp[n][j]*A[b][j] ; ct analogous with Wt/bt

__global__ void k_const(const float* __restrict__ Aans,
                        const float* __restrict__ Wp, const float* __restrict__ bp,
                        const float* __restrict__ Wt, const float* __restrict__ bt,
                        float* __restrict__ cp, float* __restrict__ ct) {
  int idx = blockIdx.x * blockDim.x + threadIdx.x;
  if (idx >= 2 * B_ * D_) return;
  int which = idx / (B_ * D_);
  int e = idx - which * (B_ * D_);
  int b = e / D_, n = e - b * D_;
  const float* W    = which ? Wt : Wp;
  const float* bias = which ? bt : bp;
  const float* av   = Aans + (size_t)(b * NL_ + (NL_ - 1)) * D_;
  const float* wr   = W + (size_t)n * (3 * D_);
  float s = bias[n];
  for (int j = 0; j < D_; ++j) s += wr[j] * av[j];
  (which ? ct : cp)[b * D_ + n] = s;
}

// ---------------- K1: big fused GEMM  [ps | ts] = X' * WT16^T ----------------
// M=24000, K=1536, N=1536. Block tile 64x128, 8 waves, 4 WMMA tiles/wave.
// K-loop split into uniform phases: kt<24 -> A tile = Q ; kt>=24 -> A tile = A*Q.
// Epilogue: n<768 -> +cp, store f16 to PS16 ; n>=768 -> +ct, atomic-max to TMAX.

__global__ __launch_bounds__(256) void k_gemm1(
    const float* __restrict__ Aans, const float* __restrict__ Q,
    const int* __restrict__ tids, const half_t* __restrict__ WT16,
    const float* __restrict__ cp, const float* __restrict__ ct,
    half_t* __restrict__ PS16, float* __restrict__ TMAX) {
  __shared__ half_t As[64 * LDA];
  __shared__ half_t Bs[128 * LDA];
  const int tid  = threadIdx.x;
  const int wave = tid >> 5, lane = tid & 31;
  const int Mbase = blockIdx.x * 64;
  const int Nbase = blockIdx.y * 128;

  // A-tile loader mapping: one v8h (8 k-values) per thread per iteration
  const int rowA = tid >> 2;                   // 0..63
  const int segA = (tid & 3) * 8;              // 0,8,16,24 halves
  const float* qptr;
  const float* aptr;
  {
    int r = Mbase + rowA;
    int b = r / T_;
    int t = r - b * T_;
    qptr = Q + ((size_t)(b * NL_ + (NL_ - 1)) * T_ + t) * D_;
    aptr = Aans + (size_t)(b * NL_ + (NL_ - 1)) * D_;
  }
  half_t* asDst = &As[rowA * LDA + segA];

  // B-tile loader mapping: two 16B chunks per thread
  const int nB   = tid >> 1;                   // 0..127
  const int segB = (tid & 1) * 16;             // halves
  const half_t* bSrcBase = WT16 + (size_t)(Nbase + nB) * K1_ + segB;
  half_t* bDst = &Bs[nB * LDA + segB];

  v8f acc[4];
#pragma unroll
  for (int j = 0; j < 4; ++j) acc[j] = (v8f)(0.0f);

  const int mrow = (wave >> 1) * 16;
  const int ncol0 = (wave & 1) * 64;

  // ---- phase 1: X' = Q ----
  for (int kt = 0; kt < D_ / 32; ++kt) {
    const int k0 = kt * 32;
    *(v8h*)asDst = cvt8(qptr + k0 + segA);
    async_cp16(bSrcBase + k0, bDst);
    async_cp16(bSrcBase + k0 + 8, bDst + 8);
    async_wait();
    __syncthreads();
    v16h a = load_a_frag(As, lane, mrow);
#pragma unroll
    for (int j = 0; j < 4; ++j)
      acc[j] = wmma_f16(a, load_b_frag(Bs, lane, ncol0 + j * 16), acc[j]);
    __syncthreads();
  }
  // ---- phase 2: X' = A * Q ----
  for (int kt = 0; kt < D_ / 32; ++kt) {
    const int k0 = kt * 32;
    f32x4 q0 = *(const f32x4*)(qptr + k0 + segA);
    f32x4 q1 = *(const f32x4*)(qptr + k0 + segA + 4);
    f32x4 a0 = *(const f32x4*)(aptr + k0 + segA);
    f32x4 a1 = *(const f32x4*)(aptr + k0 + segA + 4);
    v8h x;
    x[0] = (half_t)(q0[0] * a0[0]); x[1] = (half_t)(q0[1] * a0[1]);
    x[2] = (half_t)(q0[2] * a0[2]); x[3] = (half_t)(q0[3] * a0[3]);
    x[4] = (half_t)(q1[0] * a1[0]); x[5] = (half_t)(q1[1] * a1[1]);
    x[6] = (half_t)(q1[2] * a1[2]); x[7] = (half_t)(q1[3] * a1[3]);
    *(v8h*)asDst = x;
    async_cp16(bSrcBase + D_ + k0, bDst);
    async_cp16(bSrcBase + D_ + k0 + 8, bDst + 8);
    async_wait();
    __syncthreads();
    v16h a = load_a_frag(As, lane, mrow);
#pragma unroll
    for (int j = 0; j < 4; ++j)
      acc[j] = wmma_f16(a, load_b_frag(Bs, lane, ncol0 + j * 16), acc[j]);
    __syncthreads();
  }

  const bool isPS = (Nbase < D_);              // uniform per block
#pragma unroll
  for (int i = 0; i < 8; ++i) {
    int rl = mrow + i + ((lane >= 16) ? 8 : 0);
    int r  = Mbase + rl;
    int b  = r / T_;
    int rem = r - b * T_;
#pragma unroll
    for (int j = 0; j < 4; ++j) {
      int n = Nbase + ncol0 + j * 16 + (lane & 15);
      float v = acc[j][i];
      if (isPS) {
        v += cp[b * D_ + n];
        PS16[(size_t)r * D_ + n] = (half_t)v;
      } else {
        int nn  = n - D_;
        v += ct[b * D_ + nn];
        int p   = rem / TOK_;
        int tok = rem - p * TOK_;
        int tb  = tids[b * P_ + p];
        __hip_atomic_fetch_max(
            &TMAX[((size_t)(b * NT_ + tb) * TOK_ + tok) * D_ + nn], v,
            __ATOMIC_RELAXED, __HIP_MEMORY_SCOPE_AGENT);
      }
    }
  }
}

// ---------------- K3: G[m][n] = TMAX[m] . WA16[n] + bf1[n]  (M=2400) ----------------

__global__ __launch_bounds__(256) void k_gemm_aggr(
    const float* __restrict__ TMAX, const half_t* __restrict__ WA16,
    const float* __restrict__ bf1, float* __restrict__ G) {
  __shared__ half_t As[16 * LDA];
  __shared__ half_t Bs[128 * LDA];
  const int tid = threadIdx.x, wave = tid >> 5, lane = tid & 31;
  const int Mbase = blockIdx.x * 16;           // 150 blocks
  const int Nbase = blockIdx.y * 128;          // 6 blocks
  const int nB = tid >> 1, segB = (tid & 1) * 16;
  const half_t* bSrcBase = WA16 + (size_t)(Nbase + nB) * D_ + segB;
  half_t* bDst = &Bs[nB * LDA + segB];

  v8f acc = (v8f)(0.0f);
  for (int kt = 0; kt < D_ / 32; ++kt) {
    const int k0 = kt * 32;
    if (tid < 64) {                            // 16 rows x 32 halves, one v8h/thread
      int row = tid >> 2, seg = (tid & 3) * 8;
      *(v8h*)&As[row * LDA + seg] = cvt8(TMAX + (size_t)(Mbase + row) * D_ + k0 + seg);
    }
    async_cp16(bSrcBase + k0, bDst);
    async_cp16(bSrcBase + k0 + 8, bDst + 8);
    async_wait();
    __syncthreads();
    v16h a = load_a_frag(As, lane, 0);
    v16h b = load_b_frag(Bs, lane, wave * 16);
    acc = wmma_f16(a, b, acc);
    __syncthreads();
  }
#pragma unroll
  for (int i = 0; i < 8; ++i) {
    int m = Mbase + i + ((lane >= 16) ? 8 : 0);
    int n = Nbase + wave * 16 + (lane & 15);
    G[(size_t)m * D_ + n] = acc[i] + bf1[n];
  }
}

// ---------------- K4: stage-2 GEMM + fused relu/w2-dot epilogue ----------------
// score[r] += sum_n relu(PS16[r].WF16[n] + G[grow(r)][n]) * w2[n]

__global__ __launch_bounds__(256) void k_gemm2(
    const half_t* __restrict__ PS16, const half_t* __restrict__ WF16,
    const float* __restrict__ G, const float* __restrict__ w2,
    const int* __restrict__ tids, float* __restrict__ score) {
  __shared__ half_t As[64 * LDA];
  __shared__ half_t Bs[128 * LDA];
  const int tid = threadIdx.x, wave = tid >> 5, lane = tid & 31;
  const int Mbase = blockIdx.x * 64;           // 375 blocks
  const int Nbase = blockIdx.y * 128;          // 6 blocks
  const int rowA = tid >> 2, segA = (tid & 3) * 8;
  const int nB = tid >> 1,  segB = (tid & 1) * 16;
  const half_t* aSrcBase = PS16 + (size_t)(Mbase + rowA) * D_ + segA;
  const half_t* bSrcBase = WF16 + (size_t)(Nbase + nB) * D_ + segB;
  half_t* aDst = &As[rowA * LDA + segA];
  half_t* bDst = &Bs[nB * LDA + segB];

  v8f acc[4];
#pragma unroll
  for (int j = 0; j < 4; ++j) acc[j] = (v8f)(0.0f);

  const int mrow = (wave >> 1) * 16;
  const int ncol0 = (wave & 1) * 64;

  for (int kt = 0; kt < D_ / 32; ++kt) {
    const int k0 = kt * 32;
    async_cp16(aSrcBase + k0, aDst);           // A and B tiles are raw f16 copies
    async_cp16(bSrcBase + k0, bDst);
    async_cp16(bSrcBase + k0 + 8, bDst + 8);
    async_wait();
    __syncthreads();
    v16h a = load_a_frag(As, lane, mrow);
#pragma unroll
    for (int j = 0; j < 4; ++j)
      acc[j] = wmma_f16(a, load_b_frag(Bs, lane, ncol0 + j * 16), acc[j]);
    __syncthreads();
  }

  float rs[8];
#pragma unroll
  for (int i = 0; i < 8; ++i) {
    int rl  = mrow + i + ((lane >= 16) ? 8 : 0);
    int r   = Mbase + rl;
    int b   = r / T_;
    int rem = r - b * T_;
    int p   = rem / TOK_;
    int tok = rem - p * TOK_;
    int tb  = tids[b * P_ + p];
    size_t grow = ((size_t)(b * NT_ + tb) * TOK_ + tok) * D_;
    float s = 0.0f;
#pragma unroll
    for (int j = 0; j < 4; ++j) {
      int n = Nbase + ncol0 + j * 16 + (lane & 15);
      float v = acc[j][i] + G[grow + n];
      v = v > 0.0f ? v : 0.0f;
      s += v * w2[n];
    }
    rs[i] = s;
  }
  // reduce over the 16 lanes that share each row (wave32 xor shuffle)
#pragma unroll
  for (int i = 0; i < 8; ++i) {
    float v = rs[i];
    v += __shfl_xor(v, 1);
    v += __shfl_xor(v, 2);
    v += __shfl_xor(v, 4);
    v += __shfl_xor(v, 8);
    rs[i] = v;
  }
  if ((lane & 15) == 0) {
#pragma unroll
    for (int i = 0; i < 8; ++i) {
      int r = Mbase + mrow + i + ((lane >= 16) ? 8 : 0);
      atomicAdd(&score[r], rs[i]);
    }
  }
}

// ---------------- K5: out[b][p] = sum_tok mask * (score + bf2) ----------------

__global__ void k_final(const float* __restrict__ score, const float* __restrict__ mask,
                        const float* __restrict__ bf2, float* __restrict__ out) {
  int idx = blockIdx.x * blockDim.x + threadIdx.x;
  if (idx >= B_ * P_) return;
  const float b2 = bf2[0];
  const float* ms = mask + (size_t)idx * TOK_;
  const float* sc = score + (size_t)idx * TOK_;
  float s = 0.0f;
  for (int t = 0; t < TOK_; ++t) s += ms[t] * (sc[t] + b2);
  out[idx] = s;
}

// ---------------- launch ----------------

extern "C" void kernel_launch(void* const* d_in, const int* in_sizes, int n_in,
                              void* d_out, int out_size, void* d_ws, size_t ws_size,
                              hipStream_t stream) {
  (void)in_sizes; (void)n_in; (void)out_size; (void)ws_size;
  const float* Aans = (const float*)d_in[0];
  const float* Q    = (const float*)d_in[1];
  const float* mask = (const float*)d_in[2];
  const int*   tids = (const int*)d_in[3];
  const float* Wp   = (const float*)d_in[4];
  const float* bp   = (const float*)d_in[5];
  const float* Wt   = (const float*)d_in[6];
  const float* bt   = (const float*)d_in[7];
  const float* Wf1  = (const float*)d_in[8];
  const float* bf1  = (const float*)d_in[9];
  const float* Wf2  = (const float*)d_in[10];
  const float* bf2  = (const float*)d_in[11];
  float* out = (float*)d_out;

  // workspace layout (all offsets 256B-aligned)
  char* ws = (char*)d_ws;
  float*  TMAX  = (float*)(ws + 0);                         //  7,372,800 B
  float*  score = (float*)(ws + 7372800);                   //     96,000 B
  float*  G     = (float*)(ws + 7468800);                   //  7,372,800 B
  half_t* PS16  = (half_t*)(ws + 14841600);                 // 36,864,000 B
  half_t* WT16  = (half_t*)(ws + 51705600);                 //  4,718,592 B
  half_t* WF16  = (half_t*)(ws + 56424192);                 //  1,179,648 B
  half_t* WA16  = (half_t*)(ws + 57603840);                 //  1,179,648 B
  float*  cp    = (float*)(ws + 58783488);                  //     12,288 B
  float*  ct    = (float*)(ws + 58795776);                  //     12,288 B

  const int NTM = B_ * NT_ * TOK_ * D_;
  k_init<<<(NTM + MTOT + 255) / 256, 256, 0, stream>>>(TMAX, score);
  k_prep_w<<<(K1_ * K1_ + 2 * D_ * D_ + 255) / 256, 256, 0, stream>>>(
      Wp, Wt, Wf1, WT16, WF16, WA16);
  k_const<<<(2 * B_ * D_ + 255) / 256, 256, 0, stream>>>(Aans, Wp, bp, Wt, bt, cp, ct);
  k_gemm1<<<dim3(MTOT / 64, K1_ / 128), 256, 0, stream>>>(
      Aans, Q, tids, WT16, cp, ct, PS16, TMAX);
  k_gemm_aggr<<<dim3((B_ * NT_ * TOK_) / 16, D_ / 128), 256, 0, stream>>>(
      TMAX, WA16, bf1, G);
  k_gemm2<<<dim3(MTOT / 64, D_ / 128), 256, 0, stream>>>(
      PS16, WF16, G, Wf2, tids, score);
  k_final<<<(B_ * P_ + 255) / 256, 256, 0, stream>>>(score, mask, bf2, out);
}
